// PD_Block_32701880992218
// MI455X (gfx1250) — matrix operand order repaired and checked by reading, not compiled, not run on hardware
//
#include <hip/hip_runtime.h>

#define DEVFN __device__ __forceinline__

typedef __attribute__((ext_vector_type(16))) __bf16        v16bf;
typedef __attribute__((ext_vector_type(2)))  __bf16        v2bf;
typedef __attribute__((ext_vector_type(8)))  float         v8f;
typedef __attribute__((ext_vector_type(8)))  unsigned int  v8u;

// Problem dims (fixed by the reference)
constexpr int B_  = 8;
constexpr int CIN = 256;
constexpr int CD  = 128;
constexpr int NSP = 4096;   // 64*64

#if __has_builtin(__builtin_amdgcn_exp2f)
#define EXP2F __builtin_amdgcn_exp2f
#else
#define EXP2F exp2f
#endif

// ---------------- bf16 helpers (hardware cvt, v_cvt_pk_bf16_f32 path) ----------------
DEVFN unsigned int packbf2(float a, float b) {
  v2bf t;
  t[0] = (__bf16)a;
  t[1] = (__bf16)b;
  return __builtin_bit_cast(unsigned int, t);
}
DEVFN unsigned short bf16of(float a) {
  __bf16 t = (__bf16)a;
  return __builtin_bit_cast(unsigned short, t);
}
DEVFN float bf_lo(unsigned int u) { return __uint_as_float(u << 16); }
DEVFN float bf_hi(unsigned int u) { return __uint_as_float(u & 0xFFFF0000u); }

// ---------------- WMMA wrapper: D = A(16x32 bf16) * B(32x16 bf16) + C(f32) ----------------
DEVFN v8f wmma_bf16(v8u a, v8u b, v8f c) {
  return __builtin_amdgcn_wmma_f32_16x16x32_bf16(
      false, __builtin_bit_cast(v16bf, a),
      false, __builtin_bit_cast(v16bf, b),
      (short)0, c, false, false);
}

// 16 consecutive bf16 (32 B) -> v8u (two b128 loads)
DEVFN v8u ld_frag16(const unsigned short* p) {
  const uint4* q = reinterpret_cast<const uint4*>(p);
  uint4 t0 = q[0], t1 = q[1];
  v8u f; f[0]=t0.x; f[1]=t0.y; f[2]=t0.z; f[3]=t0.w;
         f[4]=t1.x; f[5]=t1.y; f[6]=t1.z; f[7]=t1.w;
  return f;
}
// 8 bf16 at p0 and 8 bf16 at p1 -> v8u
DEVFN v8u ld_frag_2x8(const unsigned short* p0, const unsigned short* p1) {
  uint4 t0 = *reinterpret_cast<const uint4*>(p0);
  uint4 t1 = *reinterpret_cast<const uint4*>(p1);
  v8u f; f[0]=t0.x; f[1]=t0.y; f[2]=t0.z; f[3]=t0.w;
         f[4]=t1.x; f[5]=t1.y; f[6]=t1.z; f[7]=t1.w;
  return f;
}

// =====================================================================
// proj_nd: out[b][n][d] = sum_c W[d][c] * X[b][c][n] + bias[d]   (bf16 out, [N][CD])
// WMMA form: D(M=d,N=n) = A(W rows) x B(X columns)
// =====================================================================
__global__ __launch_bounds__(128) void proj_nd_kernel(
    const float* __restrict__ X, const float* __restrict__ W,
    const float* __restrict__ bias, unsigned short* __restrict__ out)
{
  const int lane = threadIdx.x & 31;
  const int ll = lane & 15, lh = lane >> 4;
  const int w  = blockIdx.x * 4 + (threadIdx.x >> 5);
  const int b  = w >> 8;
  const int n0 = (w & 255) << 4;
  const float* Xb = X + (size_t)b * CIN * NSP;

  v8f acc[8];
  #pragma unroll
  for (int i = 0; i < 8; ++i) acc[i] = v8f{};

  #pragma unroll 1
  for (int c0 = 0; c0 < CIN; c0 += 32) {
    // B fragment: lane = column n0+ll, K rows c = c0+16*lh .. +15 (stride NSP)
    v8u bx;
    const float* xc = Xb + (size_t)(c0 + 16 * lh) * NSP + (n0 + ll);
    #pragma unroll
    for (int j = 0; j < 8; ++j)
      bx[j] = packbf2(xc[(size_t)(2 * j) * NSP], xc[(size_t)(2 * j + 1) * NSP]);

    #pragma unroll
    for (int dt = 0; dt < 8; ++dt) {
      const float* Wp = W + (size_t)(dt * 16 + ll) * CIN + c0 + 16 * lh;
      float4 w0 = *(const float4*)(Wp);
      float4 w1 = *(const float4*)(Wp + 4);
      float4 w2 = *(const float4*)(Wp + 8);
      float4 w3 = *(const float4*)(Wp + 12);
      v8u aw;
      aw[0]=packbf2(w0.x,w0.y); aw[1]=packbf2(w0.z,w0.w);
      aw[2]=packbf2(w1.x,w1.y); aw[3]=packbf2(w1.z,w1.w);
      aw[4]=packbf2(w2.x,w2.y); aw[5]=packbf2(w2.z,w2.w);
      aw[6]=packbf2(w3.x,w3.y); aw[7]=packbf2(w3.z,w3.w);
      acc[dt] = wmma_bf16(aw, bx, acc[dt]);
    }
  }
  // C layout: lane holds n = n0+ll, d = dt*16 + 8*lh + r  (r consecutive -> b128 store)
  unsigned short* orow = out + (size_t)b * NSP * CD + (size_t)(n0 + ll) * CD;
  #pragma unroll
  for (int dt = 0; dt < 8; ++dt) {
    const float* bp = bias + dt * 16 + 8 * lh;
    float4 b0 = *(const float4*)bp;
    float4 b1 = *(const float4*)(bp + 4);
    uint4 st;
    st.x = packbf2(acc[dt][0] + b0.x, acc[dt][1] + b0.y);
    st.y = packbf2(acc[dt][2] + b0.z, acc[dt][3] + b0.w);
    st.z = packbf2(acc[dt][4] + b1.x, acc[dt][5] + b1.y);
    st.w = packbf2(acc[dt][6] + b1.z, acc[dt][7] + b1.w);
    *reinterpret_cast<uint4*>(orow + dt * 16 + 8 * lh) = st;
  }
}

// =====================================================================
// proj_dn: out[b][d][n] = sum_c W[d][c] * X[b][c][n] + bias[d]   (bf16 out, [CD][N])
// WMMA form: D(M=n,N=d) = A(X^T rows, strided) x B(W^T columns, contiguous)
// =====================================================================
__global__ __launch_bounds__(128) void proj_dn_kernel(
    const float* __restrict__ X, const float* __restrict__ W,
    const float* __restrict__ bias, unsigned short* __restrict__ out)
{
  const int lane = threadIdx.x & 31;
  const int ll = lane & 15, lh = lane >> 4;
  const int w  = blockIdx.x * 4 + (threadIdx.x >> 5);
  const int b  = w >> 8;
  const int n0 = (w & 255) << 4;
  const float* Xb = X + (size_t)b * CIN * NSP;

  v8f acc[8];
  #pragma unroll
  for (int i = 0; i < 8; ++i) acc[i] = v8f{};

  #pragma unroll 1
  for (int c0 = 0; c0 < CIN; c0 += 32) {
    // A fragment: lane = row n0+ll; K = c0+8*lh+{0..7} then c0+16+8*lh+{0..7}
    v8u ax;
    const float* xc1 = Xb + (size_t)(c0 + 8 * lh) * NSP + (n0 + ll);
    const float* xc2 = xc1 + (size_t)16 * NSP;
    #pragma unroll
    for (int j = 0; j < 4; ++j) {
      ax[j]     = packbf2(xc1[(size_t)(2*j) * NSP], xc1[(size_t)(2*j+1) * NSP]);
      ax[4 + j] = packbf2(xc2[(size_t)(2*j) * NSP], xc2[(size_t)(2*j+1) * NSP]);
    }
    #pragma unroll
    for (int dt = 0; dt < 8; ++dt) {
      const float* Wp = W + (size_t)(dt * 16 + ll) * CIN + c0 + 8 * lh;
      float4 w0 = *(const float4*)(Wp);
      float4 w1 = *(const float4*)(Wp + 4);
      float4 w2 = *(const float4*)(Wp + 16);
      float4 w3 = *(const float4*)(Wp + 20);
      v8u bw;
      bw[0]=packbf2(w0.x,w0.y); bw[1]=packbf2(w0.z,w0.w);
      bw[2]=packbf2(w1.x,w1.y); bw[3]=packbf2(w1.z,w1.w);
      bw[4]=packbf2(w2.x,w2.y); bw[5]=packbf2(w2.z,w2.w);
      bw[6]=packbf2(w3.x,w3.y); bw[7]=packbf2(w3.z,w3.w);
      acc[dt] = wmma_bf16(ax, bw, acc[dt]);
    }
  }
  // C layout: lane holds d = dt*16+ll, n = n0 + 8*lh + r (consecutive -> b128 store)
  unsigned short* ob = out + (size_t)b * CD * NSP;
  #pragma unroll
  for (int dt = 0; dt < 8; ++dt) {
    float bv = bias[dt * 16 + ll];
    uint4 st;
    st.x = packbf2(acc[dt][0] + bv, acc[dt][1] + bv);
    st.y = packbf2(acc[dt][2] + bv, acc[dt][3] + bv);
    st.z = packbf2(acc[dt][4] + bv, acc[dt][5] + bv);
    st.w = packbf2(acc[dt][6] + bv, acc[dt][7] + bv);
    *reinterpret_cast<uint4*>(ob + (size_t)(dt * 16 + ll) * NSP + n0 + 8 * lh) = st;
  }
}

// =====================================================================
// flash_attn: per wave = 16 query rows (n0..n0+15) of batch b.
//   S^T = K Q^T  (C-layout of S^T == A-layout of P, no shuffles needed)
//   online softmax (base-2), O = P V^T, then Z = O/l + Tproj  -> [N][CD] bf16
// =====================================================================
__global__ __launch_bounds__(128) void flash_attn_kernel(
    const unsigned short* __restrict__ Qb,  // [B][N][CD]
    const unsigned short* __restrict__ Kb,  // [B][N][CD]
    const unsigned short* __restrict__ Vb,  // [B][CD][N]
    const unsigned short* __restrict__ Tb,  // [B][CD][N]
    unsigned short* __restrict__ Zb)        // [B][N][CD]
{
  __shared__ __align__(16) float red[4][16];
  const int lane = threadIdx.x & 31;
  const int ll = lane & 15, lh = lane >> 4;
  const int wv = threadIdx.x >> 5;
  const int w  = blockIdx.x * 4 + wv;
  const int b  = w >> 8;
  const int n0 = (w & 255) << 4;

  const unsigned short* Q  = Qb + (size_t)b * NSP * CD;
  const unsigned short* Kp = Kb + (size_t)b * NSP * CD;
  const unsigned short* Vp = Vb + (size_t)b * CD * NSP;

  // Q B-fragments for all of Cd (held across the whole m loop)
  v8u qf[4];
  #pragma unroll
  for (int q = 0; q < 4; ++q)
    qf[q] = ld_frag16(Q + (size_t)(n0 + ll) * CD + q * 32 + 16 * lh);

  v8f oacc[8];
  #pragma unroll
  for (int i = 0; i < 8; ++i) oacc[i] = v8f{};
  float row_max = -1e30f, row_sum = 0.f;
  const float KSC = 0.08838834764831845f * 1.4426950408889634f; // 1/sqrt(128) * log2(e)

  #pragma unroll 1
  for (int m0 = 0; m0 < NSP; m0 += 32) {
    // Prefetch next m-block of K (rows) and a sampling of V rows.
    if (m0 + 32 < NSP) {
      __builtin_prefetch(Kp + (size_t)(m0 + 32 + lane) * CD, 0, 3);
      __builtin_prefetch(Kp + (size_t)(m0 + 32 + lane) * CD + 64, 0, 3);
      __builtin_prefetch(Vp + (size_t)(lane) * NSP + m0 + 32, 0, 3);
      __builtin_prefetch(Vp + (size_t)(lane + 32) * NSP + m0 + 32, 0, 3);
      __builtin_prefetch(Vp + (size_t)(lane + 64) * NSP + m0 + 32, 0, 3);
      __builtin_prefetch(Vp + (size_t)(lane + 96) * NSP + m0 + 32, 0, 3);
    }
    // ---- GEMM1: two S^T tiles (m rows m0..+15 and m0+16..+31) ----
    v8f s0 = v8f{}, s1 = v8f{};
    #pragma unroll
    for (int q = 0; q < 4; ++q) {
      const unsigned short* k0 = Kp + (size_t)(m0 + ll) * CD + q * 32 + 16 * lh;
      s0 = wmma_bf16(ld_frag_2x8(k0, k0 + 8), qf[q], s0);
      const unsigned short* k1 = Kp + (size_t)(m0 + 16 + ll) * CD + q * 32 + 16 * lh;
      s1 = wmma_bf16(ld_frag_2x8(k1, k1 + 8), qf[q], s1);
    }
    // ---- online softmax over the 32 m's (row n = ll; partner lane = lane^16) ----
    float lm = -1e30f;
    #pragma unroll
    for (int r = 0; r < 8; ++r) {
      s0[r] *= KSC; s1[r] *= KSC;
      lm = fmaxf(lm, fmaxf(s0[r], s1[r]));
    }
    lm = fmaxf(lm, __shfl_xor(lm, 16, 32));
    float nm = fmaxf(row_max, lm);
    float ps = 0.f;
    #pragma unroll
    for (int r = 0; r < 8; ++r) {
      s0[r] = EXP2F(s0[r] - nm); ps += s0[r];
      s1[r] = EXP2F(s1[r] - nm); ps += s1[r];
    }
    ps += __shfl_xor(ps, 16, 32);

    // Rescale only when some row's max actually moved (wave-uniform branch;
    // when nm == row_max, alpha == 1 exactly, so skipping is bit-exact).
    if (__ballot(nm > row_max)) {
      float alpha = EXP2F(row_max - nm);
      row_sum = row_sum * alpha + ps;
      row_max = nm;
      if (lane < 16) red[wv][ll] = alpha;
      __builtin_amdgcn_wave_barrier();
      float4 a0 = *(const float4*)&red[wv][8 * lh];
      float4 a1 = *(const float4*)&red[wv][8 * lh + 4];
      float al[8] = {a0.x, a0.y, a0.z, a0.w, a1.x, a1.y, a1.z, a1.w};
      #pragma unroll
      for (int t = 0; t < 8; ++t)
        #pragma unroll
        for (int r = 0; r < 8; ++r) oacc[t][r] *= al[r];
    } else {
      row_sum += ps;
    }

    // P -> A fragment (direct register repack; layouts match)
    v8u pf;
    #pragma unroll
    for (int j = 0; j < 4; ++j) {
      pf[j]     = packbf2(s0[2*j], s0[2*j+1]);
      pf[4 + j] = packbf2(s1[2*j], s1[2*j+1]);
    }
    // ---- GEMM2: O(n x c) += P(n x 32m) * V^T(32m x c) ----
    #pragma unroll
    for (int ct = 0; ct < 8; ++ct) {
      const unsigned short* vp = Vp + (size_t)(ct * 16 + ll) * NSP + m0 + 8 * lh;
      oacc[ct] = wmma_bf16(pf, ld_frag_2x8(vp, vp + 16), oacc[ct]);
    }
  }

  // ---- finalize: Z = O / l + Tproj ----
  float inv = 1.0f / row_sum;
  if (lane < 16) red[wv][ll] = inv;
  __builtin_amdgcn_wave_barrier();
  float4 i0 = *(const float4*)&red[wv][8 * lh];
  float4 i1 = *(const float4*)&red[wv][8 * lh + 4];
  float iv[8] = {i0.x, i0.y, i0.z, i0.w, i1.x, i1.y, i1.z, i1.w};

  const unsigned short* Tp = Tb + (size_t)b * CD * NSP;
  unsigned short* Z = Zb + (size_t)b * NSP * CD;
  #pragma unroll
  for (int ct = 0; ct < 8; ++ct) {
    uint4 tt = *(const uint4*)(Tp + (size_t)(ct * 16 + ll) * NSP + n0 + 8 * lh);
    float tv[8] = {bf_lo(tt.x), bf_hi(tt.x), bf_lo(tt.y), bf_hi(tt.y),
                   bf_lo(tt.z), bf_hi(tt.z), bf_lo(tt.w), bf_hi(tt.w)};
    #pragma unroll
    for (int r = 0; r < 8; ++r) {
      float zv = oacc[ct][r] * iv[r] + tv[r];
      int n = n0 + 8 * lh + r;
      Z[(size_t)n * CD + ct * 16 + ll] = bf16of(zv);
    }
  }
}

// =====================================================================
// out_proj: out[b][o][n] = sum_d Wf[o][d] * Z[b][n][d] + bf[o]   (f32 out)
// WMMA form: D(M=n,N=o) = A(Z rows) x B(Wf^T columns)
// =====================================================================
__global__ __launch_bounds__(128) void out_proj_kernel(
    const unsigned short* __restrict__ Zb, const float* __restrict__ Wf,
    const float* __restrict__ bf, float* __restrict__ out)
{
  const int lane = threadIdx.x & 31;
  const int ll = lane & 15, lh = lane >> 4;
  const int w  = blockIdx.x * 4 + (threadIdx.x >> 5);
  const int b  = w >> 8;
  const int n0 = (w & 255) << 4;
  const unsigned short* Z = Zb + (size_t)b * NSP * CD;

  v8f acc[16];
  #pragma unroll
  for (int i = 0; i < 16; ++i) acc[i] = v8f{};

  #pragma unroll 1
  for (int c0 = 0; c0 < CD; c0 += 32) {
    const unsigned short* zp = Z + (size_t)(n0 + ll) * CD + c0 + 8 * lh;
    v8u az = ld_frag_2x8(zp, zp + 16);
    #pragma unroll
    for (int ot = 0; ot < 16; ++ot) {
      const float* Wp = Wf + (size_t)(ot * 16 + ll) * CD + c0 + 8 * lh;
      float4 w0 = *(const float4*)(Wp);
      float4 w1 = *(const float4*)(Wp + 4);
      float4 w2 = *(const float4*)(Wp + 16);
      float4 w3 = *(const float4*)(Wp + 20);
      v8u bw;
      bw[0]=packbf2(w0.x,w0.y); bw[1]=packbf2(w0.z,w0.w);
      bw[2]=packbf2(w1.x,w1.y); bw[3]=packbf2(w1.z,w1.w);
      bw[4]=packbf2(w2.x,w2.y); bw[5]=packbf2(w2.z,w2.w);
      bw[6]=packbf2(w3.x,w3.y); bw[7]=packbf2(w3.z,w3.w);
      acc[ot] = wmma_bf16(az, bw, acc[ot]);
    }
  }
  // C layout: lane holds o = ot*16+ll, n = n0 + 8*lh + r  -> two b128 f32 stores
  float* ob = out + (size_t)b * CIN * NSP;
  #pragma unroll
  for (int ot = 0; ot < 16; ++ot) {
    float bv = bf[ot * 16 + ll];
    float* op = ob + (size_t)(ot * 16 + ll) * NSP + n0 + 8 * lh;
    float4 s0 = {acc[ot][0] + bv, acc[ot][1] + bv, acc[ot][2] + bv, acc[ot][3] + bv};
    float4 s1 = {acc[ot][4] + bv, acc[ot][5] + bv, acc[ot][6] + bv, acc[ot][7] + bv};
    *(float4*)op = s0;
    *(float4*)(op + 4) = s1;
  }
}

// =====================================================================
extern "C" void kernel_launch(void* const* d_in, const int* in_sizes, int n_in,
                              void* d_out, int out_size, void* d_ws, size_t ws_size,
                              hipStream_t stream) {
  const float* feat_t = (const float*)d_in[0];
  const float* feat_i = (const float*)d_in[1];
  const float* Wq  = (const float*)d_in[2];
  const float* bq  = (const float*)d_in[3];
  const float* Wk  = (const float*)d_in[4];
  const float* bk  = (const float*)d_in[5];
  const float* Wv  = (const float*)d_in[6];
  const float* bv  = (const float*)d_in[7];
  const float* Wt  = (const float*)d_in[8];
  const float* bt  = (const float*)d_in[9];
  const float* Wf  = (const float*)d_in[10];
  const float* bfb = (const float*)d_in[11];
  float* out = (float*)d_out;

  // Workspace: 5 bf16 tensors of B*N*CD = 8 MiB each (40 MiB total)
  const size_t tile = (size_t)B_ * NSP * CD;
  unsigned short* Qb = (unsigned short*)d_ws;
  unsigned short* Kb = Qb + tile;
  unsigned short* Vb = Kb + tile;
  unsigned short* Tb = Vb + tile;
  unsigned short* Zb = Tb + tile;

  dim3 grid(512), block(128); // 4 wave32 per block, 1 wave = one 16-row n-tile

  hipLaunchKernelGGL(proj_nd_kernel, grid, block, 0, stream, feat_t, Wq, bq, Qb);
  hipLaunchKernelGGL(proj_nd_kernel, grid, block, 0, stream, feat_i, Wk, bk, Kb);
  hipLaunchKernelGGL(proj_dn_kernel, grid, block, 0, stream, feat_i, Wv, bv, Vb);
  hipLaunchKernelGGL(proj_dn_kernel, grid, block, 0, stream, feat_t, Wt, bt, Tb);
  hipLaunchKernelGGL(flash_attn_kernel, grid, block, 0, stream, Qb, Kb, Vb, Tb, Zb);
  hipLaunchKernelGGL(out_proj_kernel, grid, block, 0, stream, Zb, Wf, bfb, out);
}
